// VisionAttention_16338055594408
// MI455X (gfx1250) — matrix operand order repaired and checked by reading, not compiled, not run on hardware
//
#include <hip/hip_runtime.h>
#include <hip/hip_bf16.h>

typedef __attribute__((ext_vector_type(16))) __bf16 v16bf;
typedef __attribute__((ext_vector_type(8)))  __bf16 v8bf;
typedef __attribute__((ext_vector_type(8)))  float  v8f;

#define NT   4096   // tokens
#define NTP  4128   // token capacity padded for partial 32-key blocks
#define NH   16     // heads
#define HD   80     // head dim
#define HDP  96     // head dim padded to 3x32 bf16 WMMA K-chunks
#define VDP  96     // V dim rows padded to 96 for uniform async copy counts
#define DIM  1280

static __device__ __forceinline__ v16bf join8(v8bf lo, v8bf hi) {
  return __builtin_shufflevector(lo, hi, 0,1,2,3,4,5,6,7,8,9,10,11,12,13,14,15);
}
static __device__ __forceinline__ v8f wmma_bf16(v16bf a, v16bf b, v8f c) {
  return __builtin_amdgcn_wmma_f32_16x16x32_bf16(false, a, false, b, (short)0, c,
                                                 false, false);
}
// Async global->LDS copies (CDNA5, ASYNCcnt-tracked). GVS mode:
// mem = SGPR64 base + per-lane u32 offset; first operand = LDS byte address.
static __device__ __forceinline__ void async_b128(unsigned lds, const void* base,
                                                  unsigned voff) {
  asm volatile("global_load_async_to_lds_b128 %0, %1, %2"
               :: "v"(lds), "v"(voff), "s"(base) : "memory");
}
static __device__ __forceinline__ void async_b64(unsigned lds, const void* base,
                                                 unsigned voff) {
  asm volatile("global_load_async_to_lds_b64 %0, %1, %2"
               :: "v"(lds), "v"(voff), "s"(base) : "memory");
}
#define WAIT_ASYNC(n) asm volatile("s_wait_asynccnt %0" :: "i"(n) : "memory")
static __device__ __forceinline__ unsigned lds_addr(const void* p) {
  return (unsigned)(unsigned long long)p;
}

// ---------------------------------------------------------------------------
// One-time fp32 -> bf16 conversions.
// ---------------------------------------------------------------------------
__global__ __launch_bounds__(256) void cvt_bf16(const float* __restrict__ in,
                                                __bf16* __restrict__ out) {
  const size_t i = (size_t)blockIdx.x * blockDim.x + threadIdx.x;
  out[i] = (__bf16)in[i];
}
// Wt[n][k] = (bf16) W[k][n]  (W row-major [K][N])
__global__ __launch_bounds__(256) void transpose_cvt(const float* __restrict__ W,
                                                     __bf16* __restrict__ Wt,
                                                     int K, int N) {
  const size_t idx = (size_t)blockIdx.x * blockDim.x + threadIdx.x;
  const int n = (int)(idx / K), k = (int)(idx % K);
  Wt[idx] = (__bf16)W[(size_t)k * N + n];
}

// ---------------------------------------------------------------------------
// C[M,N] = A[M,K] (bf16) x Bt[N,K] (bf16, pre-transposed) + bias, fp32 accum.
// 128x128x32 tile, async double-buffered LDS staging, 8 waves x (2x4 tiles).
// ---------------------------------------------------------------------------
__global__ __launch_bounds__(256) void gemm_bias_bf16(
    const __bf16* __restrict__ A, const __bf16* __restrict__ Bt,
    const float* __restrict__ bias, float* __restrict__ C,
    int M, int N, int K)
{
  __shared__ __align__(16) __bf16 As[2][128][40];  // stride 40 -> conflict-free
  __shared__ __align__(16) __bf16 Bs[2][128][40];

  const int tid  = threadIdx.x;
  const int m0   = blockIdx.x * 128, n0 = blockIdx.y * 128;
  const int wave = tid >> 5, lane = tid & 31;
  const int wr   = wave >> 1, wc = wave & 1;
  const int lrow = lane & 15, hi = lane >> 4;
  const unsigned as_base = lds_addr(&As[0][0][0]);
  const unsigned bs_base = lds_addr(&Bs[0][0][0]);

  v8f acc[2][4];
#pragma unroll
  for (int mt = 0; mt < 2; ++mt)
#pragma unroll
    for (int nt = 0; nt < 4; ++nt)
#pragma unroll
      for (int r = 0; r < 8; ++r) acc[mt][nt][r] = 0.f;

  // 512 b128 chunks per 128x32 tile; each thread issues 2 per matrix (4 total).
  auto issue = [&](int buf, int k0) {
#pragma unroll
    for (int u = 0; u < 2; ++u) {
      const int cc = tid + u * 256, row = cc >> 2, part = cc & 3;
      async_b128(as_base + buf * 10240 + row * 80 + part * 16, A,
                 (unsigned)(((size_t)(m0 + row) * K + k0) * 2) + part * 16);
    }
#pragma unroll
    for (int u = 0; u < 2; ++u) {
      const int cc = tid + u * 256, row = cc >> 2, part = cc & 3;
      async_b128(bs_base + buf * 10240 + row * 80 + part * 16, Bt,
                 (unsigned)(((size_t)(n0 + row) * K + k0) * 2) + part * 16);
    }
  };

  const int ntiles = K >> 5;
  issue(0, 0);
  for (int t = 0; t < ntiles; ++t) {
    const int cur = t & 1;
    if (t + 1 < ntiles) { issue(cur ^ 1, (t + 1) * 32); WAIT_ASYNC(4); }
    else                { WAIT_ASYNC(0); }
    __syncthreads();

    v16bf af[2], bfr[4];
#pragma unroll
    for (int mt = 0; mt < 2; ++mt) {
      const __bf16* p = &As[cur][wr*32 + mt*16 + lrow][hi*8];
      af[mt] = join8(*(const v8bf*)p, *(const v8bf*)(p + 16));
    }
#pragma unroll
    for (int nt = 0; nt < 4; ++nt) {
      const __bf16* p = &Bs[cur][wc*64 + nt*16 + lrow][hi*8];
      bfr[nt] = join8(*(const v8bf*)p, *(const v8bf*)(p + 16));
    }
#pragma unroll
    for (int mt = 0; mt < 2; ++mt)
#pragma unroll
      for (int nt = 0; nt < 4; ++nt)
        acc[mt][nt] = wmma_bf16(af[mt], bfr[nt], acc[mt][nt]);
    __syncthreads();
  }

#pragma unroll
  for (int nt = 0; nt < 4; ++nt) {
    const int col = n0 + wc*64 + nt*16 + lrow;
    const float bv = bias[col];
#pragma unroll
    for (int mt = 0; mt < 2; ++mt)
#pragma unroll
      for (int r = 0; r < 8; ++r) {
        const int row = m0 + wr*32 + mt*16 + r + hi*8;
        C[(size_t)row * N + col] = acc[mt][nt][r] + bv;
      }
  }
}

// ---------------------------------------------------------------------------
// RoPE + bf16 packing. 1/sqrt(80) folded into Q. Q/K: [head][token][96],
// dims 80..95 zero. V transposed: [head][96 rows (80 used)][token 4128].
// ---------------------------------------------------------------------------
__global__ __launch_bounds__(256) void rope_pack(
    const float* __restrict__ qkv, const float* __restrict__ freqs,
    __bf16* __restrict__ Qb, __bf16* __restrict__ Kb, __bf16* __restrict__ Vt)
{
  const int idx   = blockIdx.x * blockDim.x + threadIdx.x;
  const int token = idx >> 4;
  const int head  = idx & 15;
  const float scale = 0.11180339887498949f;

  const float* base = qkv + (size_t)token * (3 * DIM) + head * HD;
  __bf16* qo = Qb + ((size_t)head * NT  + token) * HDP;
  __bf16* ko = Kb + ((size_t)head * NTP + token) * HDP;

#pragma unroll 4
  for (int d = 0; d < 40; ++d) {
    const float fr = freqs[(size_t)token * 40 + d];
    const float cs = __cosf(fr), sn = __sinf(fr);
    const float q1 = base[d],       q2 = base[d + 40];
    const float k1 = base[DIM + d], k2 = base[DIM + d + 40];
    qo[d]      = (__bf16)((q1 * cs - q2 * sn) * scale);
    qo[d + 40] = (__bf16)((q2 * cs + q1 * sn) * scale);
    ko[d]      = (__bf16)(k1 * cs - k2 * sn);
    ko[d + 40] = (__bf16)(k2 * cs + k1 * sn);
    const float v1 = base[2 * DIM + d], v2 = base[2 * DIM + d + 40];
    Vt[((size_t)head * VDP + d)      * NTP + token] = (__bf16)v1;
    Vt[((size_t)head * VDP + d + 40) * NTP + token] = (__bf16)v2;
  }
#pragma unroll
  for (int d = 80; d < 96; ++d) { qo[d] = (__bf16)0.f; ko[d] = (__bf16)0.f; }
}

// ---------------------------------------------------------------------------
// Segmented flash attention, workgroup-shared async K/V staging.
// grid = (qblocks=11, segments=4, heads=16); 8 waves x 16 q-rows.
// ---------------------------------------------------------------------------
__global__ __launch_bounds__(256) void attn_fwd(
    const __bf16* __restrict__ Qb, const __bf16* __restrict__ Kb,
    const __bf16* __restrict__ Vt, const int* __restrict__ cu,
    __bf16* __restrict__ Out)
{
  __shared__ __align__(16) __bf16 Ks[2][32][104];  // [key][dim96 pad104]
  __shared__ __align__(16) __bf16 Vs[2][96][40];   // [dim][tok32 pad40]
  __shared__ __align__(16) __bf16 P[8][16][40];    // wave-private P panels

  const int seg = blockIdx.y, head = blockIdx.z;
  const int s0 = cu[seg], s1 = cu[seg + 1];
  const int tid = threadIdx.x;
  const int wave = tid >> 5, lane = tid & 31;
  const int lrow = lane & 15, hi = lane >> 4;
  const int qt0 = s0 + blockIdx.x * 128 + wave * 16;  // may be >= s1 (idle wave)
  const unsigned ks_base = lds_addr(&Ks[0][0][0]);
  const unsigned vs_base = lds_addr(&Vs[0][0][0]);

  int qtok = qt0 + lrow;
  if (qtok >= s1) qtok = s0;  // safe dummy row; writes guarded
  const __bf16* qp = Qb + ((size_t)(head * NT + qtok)) * HDP;
  v16bf qa[3];
#pragma unroll
  for (int c = 0; c < 3; ++c) {
    const __bf16* p = qp + c * 32 + hi * 8;
    qa[c] = join8(*(const v8bf*)p, *(const v8bf*)(p + 16));
  }

  // Per tile: K = 32 rows x 24 b64 chunks, V = 96 rows x 8 b64 chunks;
  // 768 chunks each -> every thread issues exactly 3 + 3 async ops.
  auto issue = [&](int buf, int kb) {
#pragma unroll
    for (int u = 0; u < 3; ++u) {
      const int cc = tid + u * 256, row = cc / 24, off = cc % 24;
      async_b64(ks_base + buf * 6656 + row * 208 + off * 8, Kb,
                (unsigned)((((size_t)(head * NTP + kb + row)) * HDP + off * 4) * 2));
    }
#pragma unroll
    for (int u = 0; u < 3; ++u) {
      const int cc = tid + u * 256, d = cc >> 3, c8 = cc & 7;
      async_b64(vs_base + buf * 7680 + d * 80 + c8 * 8, Vt,
                (unsigned)((((size_t)(head * VDP + d)) * NTP + kb + c8 * 4) * 2));
    }
  };

  float m8[8], l8[8];
  v8f acc[5];
#pragma unroll
  for (int r = 0; r < 8; ++r) { m8[r] = -3.0e38f; l8[r] = 0.f; }
#pragma unroll
  for (int dt = 0; dt < 5; ++dt)
#pragma unroll
    for (int r = 0; r < 8; ++r) acc[dt][r] = 0.f;

  const int nkb = (s1 - s0 + 31) >> 5;
  issue(0, s0);
  for (int j = 0; j < nkb; ++j) {
    const int kb = s0 + j * 32;
    const int cur = j & 1;
    if (j + 1 < nkb) { issue(cur ^ 1, kb + 32); WAIT_ASYNC(6); }
    else             { WAIT_ASYNC(0); }
    __syncthreads();

    // ---- S = Q K^T (scale folded into Q) ----
    v8f s[2];
#pragma unroll
    for (int nt = 0; nt < 2; ++nt) {
#pragma unroll
      for (int r = 0; r < 8; ++r) s[nt][r] = 0.f;
      const int klocal = nt * 16 + lrow;  // B-layout: lane = key column
#pragma unroll
      for (int c = 0; c < 3; ++c) {
        const __bf16* p = &Ks[cur][klocal][c * 32 + hi * 8];
        v16bf b = join8(*(const v8bf*)p, *(const v8bf*)(p + 16));
        s[nt] = wmma_bf16(qa[c], b, s[nt]);
      }
      if (kb + klocal >= s1) {
#pragma unroll
        for (int r = 0; r < 8; ++r) s[nt][r] = -3.0e38f;  // segment mask
      }
    }

    // ---- online softmax (row reductions within 16-lane halves) ----
    float t8[8];
#pragma unroll
    for (int r = 0; r < 8; ++r) t8[r] = fmaxf(s[0][r], s[1][r]);
#pragma unroll
    for (int off = 1; off < 16; off <<= 1)
#pragma unroll
      for (int r = 0; r < 8; ++r) t8[r] = fmaxf(t8[r], __shfl_xor(t8[r], off, 32));

    float f8[8], ps[8];
#pragma unroll
    for (int r = 0; r < 8; ++r) {
      const float mn = fmaxf(m8[r], t8[r]);
      f8[r] = __expf(m8[r] - mn);
      m8[r] = mn;
      ps[r] = 0.f;
    }
#pragma unroll
    for (int nt = 0; nt < 2; ++nt)
#pragma unroll
      for (int r = 0; r < 8; ++r) {
        const float p = __expf(s[nt][r] - m8[r]);
        ps[r] += p;
        P[wave][r + hi * 8][nt * 16 + lrow] = (__bf16)p;  // C-layout -> panel
      }
#pragma unroll
    for (int off = 1; off < 16; off <<= 1)
#pragma unroll
      for (int r = 0; r < 8; ++r) ps[r] += __shfl_xor(ps[r], off, 32);
#pragma unroll
    for (int r = 0; r < 8; ++r) l8[r] = l8[r] * f8[r] + ps[r];
#pragma unroll
    for (int dt = 0; dt < 5; ++dt)
#pragma unroll
      for (int r = 0; r < 8; ++r) acc[dt][r] *= f8[r];

    // ---- PV: reload P in A-layout, V from shared staging ----
    const __bf16* pp = &P[wave][lrow][hi * 8];
    v16bf pa = join8(*(const v8bf*)pp, *(const v8bf*)(pp + 16));
#pragma unroll
    for (int dt = 0; dt < 5; ++dt) {
      const __bf16* vp = &Vs[cur][dt * 16 + lrow][hi * 8];
      v16bf vb = join8(*(const v8bf*)vp, *(const v8bf*)(vp + 16));
      acc[dt] = wmma_bf16(pa, vb, acc[dt]);
    }
    __syncthreads();  // all reads of buf `cur` retired before it is re-issued
  }

#pragma unroll
  for (int dt = 0; dt < 5; ++dt) {
    const int col = head * HD + dt * 16 + lrow;
#pragma unroll
    for (int r = 0; r < 8; ++r) {
      const int tok = qt0 + r + hi * 8;
      if (tok < s1) Out[(size_t)tok * DIM + col] = (__bf16)(acc[dt][r] / l8[r]);
    }
  }
}

// ---------------------------------------------------------------------------
extern "C" void kernel_launch(void* const* d_in, const int* in_sizes, int n_in,
                              void* d_out, int out_size, void* d_ws, size_t ws_size,
                              hipStream_t stream) {
  (void)in_sizes; (void)n_in; (void)out_size; (void)ws_size;
  const float* hidden = (const float*)d_in[0];
  const int*   cu     = (const int*)  d_in[1];
  const float* freqs  = (const float*)d_in[2];
  const float* w_qkv  = (const float*)d_in[3];
  const float* b_qkv  = (const float*)d_in[4];
  const float* w_proj = (const float*)d_in[5];
  const float* b_proj = (const float*)d_in[6];
  float* out = (float*)d_out;

  char* ws = (char*)d_ws;
  const size_t OFF_QKV = 0;                                   // fp32 qkv (62,914,560)
  const size_t OFF_ABF = OFF_QKV + (size_t)NT * 3 * DIM * 4;  // bf16 A / attn-out
  const size_t OFF_WQ  = OFF_ABF + (size_t)NT * DIM * 2;      // Wqkv_t
  const size_t OFF_WP  = OFF_WQ  + (size_t)3 * DIM * DIM * 2; // Wproj_t
  const size_t OFF_QB  = OFF_WP  + (size_t)DIM * DIM * 2;
  const size_t OFF_KB  = OFF_QB  + (size_t)NH * NT  * HDP * 2;
  const size_t OFF_VT  = OFF_KB  + (size_t)NH * NTP * HDP * 2;

  float*  qkv  = (float*) (ws + OFF_QKV);
  __bf16* Abf  = (__bf16*)(ws + OFF_ABF);  // hidden bf16, later attn-out bf16
  __bf16* Wqt  = (__bf16*)(ws + OFF_WQ);
  __bf16* Wpt  = (__bf16*)(ws + OFF_WP);
  __bf16* Qb   = (__bf16*)(ws + OFF_QB);
  __bf16* Kb   = (__bf16*)(ws + OFF_KB);
  __bf16* Vt   = (__bf16*)(ws + OFF_VT);

  // 0) one-time bf16 conversions / weight transposes
  cvt_bf16<<<(NT * DIM) / 256, 256, 0, stream>>>(hidden, Abf);
  transpose_cvt<<<(3 * DIM * DIM) / 256, 256, 0, stream>>>(w_qkv, Wqt, DIM, 3 * DIM);
  transpose_cvt<<<(DIM * DIM) / 256, 256, 0, stream>>>(w_proj, Wpt, DIM, DIM);
  // 1) qkv = hidden @ w_qkv + b_qkv
  gemm_bias_bf16<<<dim3(NT / 128, (3 * DIM) / 128), 256, 0, stream>>>(
      Abf, Wqt, b_qkv, qkv, NT, 3 * DIM, DIM);
  // 2) rotary + packing
  rope_pack<<<(NT * NH) / 256, 256, 0, stream>>>(qkv, freqs, Qb, Kb, Vt);
  // 3) attention -> bf16 (overwrites Abf, which is dead after step 1)
  attn_fwd<<<dim3(11, 4, NH), 256, 0, stream>>>(Qb, Kb, Vt, cu, Abf);
  // 4) out = attn @ w_proj + b_proj
  gemm_bias_bf16<<<dim3(NT / 128, DIM / 128), 256, 0, stream>>>(
      Abf, Wpt, b_proj, out, NT, DIM, DIM);
}
